// CustomTransformerBlock_8048768713029
// MI455X (gfx1250) — matrix operand order repaired and checked by reading, not compiled, not run on hardware
//
#include <hip/hip_runtime.h>
#include <hip/hip_bf16.h>
#include <math.h>

// ---------------------------------------------------------------------------
// Types & WMMA helpers (CDNA5 / gfx1250, wave32)
// ---------------------------------------------------------------------------
typedef __bf16 bf16;
typedef __attribute__((ext_vector_type(16))) __bf16 bf16x16;
typedef __attribute__((ext_vector_type(8)))  float  f32x8;
typedef __attribute__((ext_vector_type(4)))  unsigned int u32x4;
typedef __attribute__((ext_vector_type(8)))  int i32x8;
typedef __attribute__((ext_vector_type(4)))  int i32x4;

#define SEQ    2048
#define BATCH  4
#define HDIM   2048
#define NHEADS 16
#define HEADD  128
#define MROWS  (SEQ * BATCH)   // 8192 token rows

#if defined(__HIP_DEVICE_COMPILE__) && __has_builtin(__builtin_amdgcn_tensor_load_to_lds)
#define HAS_TDM 1
#else
#define HAS_TDM 0
#endif

union Frag {                      // 32-byte bf16 fragment (8 VGPRs)
    uint4   u[2];
    bf16x16 v;
};

__device__ __forceinline__ f32x8 wmma_bf16(const bf16x16 a, const bf16x16 b, f32x8 c) {
    return __builtin_amdgcn_wmma_f32_16x16x32_bf16(
        false, a, false, b, (short)0, c, false, false);
}

// A-frag (16x32 MxK): lane<16 -> row l, K={0..7,16..23}; lane>=16 -> K={8..15,24..31}
__device__ __forceinline__ Frag ld_afrag(const bf16* base, int h) {
    Frag f;
    f.u[0] = *reinterpret_cast<const uint4*>(base + h * 8);
    f.u[1] = *reinterpret_cast<const uint4*>(base + h * 8 + 16);
    return f;
}
// B-frag (32x16 KxN): lane<16 -> col l, K={0..7,16..23}; lane>=16 -> K={8..15,24..31}
__device__ __forceinline__ Frag ld_bfrag(const bf16* base, int h) {
    Frag f;
    f.u[0] = *reinterpret_cast<const uint4*>(base + h * 8);
    f.u[1] = *reinterpret_cast<const uint4*>(base + h * 8 + 16);
    return f;
}

__device__ __forceinline__ f32x8 fzero() {
    f32x8 z = {0.f, 0.f, 0.f, 0.f, 0.f, 0.f, 0.f, 0.f};
    return z;
}

#if HAS_TDM
// Tensor Data Mover: DMA a fully in-bounds 2D tile (d0 contiguous elements per
// row, d1 rows, row pitch stride0 elements, 2-byte elements) into LDS.
// D# packing per CDNA5 ISA ch. 8 (group0: count/lds/global/type=2; group1:
// data_size=2B, tensor dims = tile dims, dim0 stride). 6-arg builtin form.
__device__ __forceinline__ void tdm_load_2d(unsigned lds_byte,
                                            const void* gptr,
                                            unsigned d0, unsigned d1,
                                            unsigned long long stride0) {
    unsigned long long ga = (unsigned long long)(size_t)gptr;
    u32x4 g0;
    g0[0] = 1u;                                     // count=1, user descriptor
    g0[1] = lds_byte;                               // lds_addr
    g0[2] = (unsigned)(ga & 0xFFFFFFFFull);         // global_addr[31:0]
    g0[3] = (unsigned)((ga >> 32) & 0x1FFFFFFull) | (2u << 30);  // [56:32] | type=2
    i32x8 g1;
    g1[0] = (int)(1u << 16);                        // data_size = 2 bytes
    g1[1] = (int)((d0 & 0xFFFFu) << 16);            // tensor_dim0[15:0]
    g1[2] = (int)((d0 >> 16) | ((d1 & 0xFFFFu) << 16));   // td0 hi | td1 lo
    g1[3] = (int)((d1 >> 16) | (d0 << 16));         // td1 hi | tile_dim0
    g1[4] = (int)(d1 & 0xFFFFu);                    // tile_dim1 (tile_dim2 = 0)
    g1[5] = (int)(stride0 & 0xFFFFFFFFull);         // tensor_dim0_stride[31:0]
    g1[6] = (int)((stride0 >> 32) & 0xFFFFull);     // stride0[47:32] (stride1=0)
    g1[7] = 0;
    i32x4 z4 = {0, 0, 0, 0};
    i32x8 z8 = {0, 0, 0, 0, 0, 0, 0, 0};
    __builtin_amdgcn_tensor_load_to_lds(g0, g1, z4, z4, z8, 0);
}
#endif

// ---------------------------------------------------------------------------
// f32 -> bf16 weight conversion (one-time)
// ---------------------------------------------------------------------------
__global__ __launch_bounds__(256) void cvt_kernel(const float* __restrict__ in,
                                                  bf16* __restrict__ out, int n) {
    int i = (blockIdx.x * 256 + threadIdx.x) * 4;
    if (i < n) {
        float4 f = *reinterpret_cast<const float4*>(in + i);
        out[i + 0] = (bf16)f.x; out[i + 1] = (bf16)f.y;
        out[i + 2] = (bf16)f.z; out[i + 3] = (bf16)f.w;
    }
}

// ---------------------------------------------------------------------------
// LayerNorm -> bf16 cast (one block per token row, H = 2048)
// ---------------------------------------------------------------------------
__global__ __launch_bounds__(256) void ln_kernel(
    const float* __restrict__ x, const float* __restrict__ g,
    const float* __restrict__ b, bf16* __restrict__ y, int H)
{
    __shared__ float red[8];
    const size_t row = blockIdx.x;
    const float* xr = x + row * (size_t)H;
    const int tid = threadIdx.x;

    float s = 0.f;
    for (int i = tid; i < H; i += 256) s += xr[i];
    for (int o = 16; o > 0; o >>= 1) s += __shfl_xor(s, o, 32);
    if ((tid & 31) == 0) red[tid >> 5] = s;
    __syncthreads();
    float tot = 0.f;
    for (int w = 0; w < 8; w++) tot += red[w];
    const float mu = tot / (float)H;
    __syncthreads();

    float v = 0.f;
    for (int i = tid; i < H; i += 256) { float d = xr[i] - mu; v += d * d; }
    for (int o = 16; o > 0; o >>= 1) v += __shfl_xor(v, o, 32);
    if ((tid & 31) == 0) red[tid >> 5] = v;
    __syncthreads();
    float vt = 0.f;
    for (int w = 0; w < 8; w++) vt += red[w];
    const float rstd = rsqrtf(vt / (float)H + 1e-5f);

    bf16* yr = y + row * (size_t)H;
    for (int i = tid; i < H; i += 256)
        yr[i] = (bf16)((xr[i] - mu) * rstd * g[i] + b[i]);
}

// ---------------------------------------------------------------------------
// GEMM: C[M,N] = A[M,K](bf16) * W[N,K](bf16)^T + bias, block tile 128x128x64,
// TDM double-buffered LDS staging, 8 waves (2m x 4n), 16 WMMA / wave / step.
// EPI: 0 -> bf16 out; 1 -> f32 out + residual; 2 -> exact GELU -> bf16 out.
// ---------------------------------------------------------------------------
template <int EPI>
__global__ __launch_bounds__(256) void gemm_kernel(
    const bf16*  __restrict__ A, const bf16* __restrict__ W,
    const float* __restrict__ bias, const float* __restrict__ resid,
    void* __restrict__ out, int M, int N, int K)
{
    // [As0 | As1 | Bs0 | Bs1], each 128x64 bf16 = 16 KB (64 KB total)
    __shared__ bf16 smem[4 * 128 * 64];
    const unsigned ASZ = 128 * 64;

    const int tid  = threadIdx.x;
    const int lane = tid & 31, wid = tid >> 5;
    const int l = lane & 15, h = lane >> 4;
    const int wm = wid & 1, wn = wid >> 1;          // 2 x 4 wave grid
    const int m0 = blockIdx.y * 128, n0 = blockIdx.x * 128;

    f32x8 acc[4][2];
    for (int i = 0; i < 4; i++)
        for (int j = 0; j < 2; j++) acc[i][j] = fzero();

#if HAS_TDM
    // Prologue: DMA tile 0 into buffer 0 (wave 0 issues; TENSORcnt tracks).
    if (wid == 0) {
        tdm_load_2d(0 * ASZ * 2, &A[(size_t)m0 * K], 64, 128, (unsigned)K);
        tdm_load_2d(2 * ASZ * 2, &W[(size_t)n0 * K], 64, 128, (unsigned)K);
    }
    int p = 0;
    for (int k0 = 0; k0 < K; k0 += 64) {
        __syncthreads();                       // prior compute done everywhere
        const bool more = (k0 + 64) < K;
        if (wid == 0) {
            if (more) {
                tdm_load_2d((unsigned)((1 - p) * ASZ * 2),
                            &A[(size_t)m0 * K + k0 + 64], 64, 128, (unsigned)K);
                tdm_load_2d((unsigned)((2 + (1 - p)) * ASZ * 2),
                            &W[(size_t)n0 * K + k0 + 64], 64, 128, (unsigned)K);
                __builtin_amdgcn_s_wait_tensorcnt(2);   // current tile resident
            } else {
                __builtin_amdgcn_s_wait_tensorcnt(0);
            }
        }
        __syncthreads();                       // current buffer visible to all

        const bf16* Ab = smem + p * ASZ;
        const bf16* Bb = smem + (2 + p) * ASZ;
        for (int kk = 0; kk < 2; kk++) {
            Frag a[4], bfr[2];
            for (int i = 0; i < 4; i++)
                a[i] = ld_afrag(&Ab[(wm * 64 + i * 16 + l) * 64 + kk * 32], h);
            for (int j = 0; j < 2; j++)
                bfr[j] = ld_bfrag(&Bb[(wn * 32 + j * 16 + l) * 64 + kk * 32], h);
            for (int i = 0; i < 4; i++)
                for (int j = 0; j < 2; j++)
                    acc[i][j] = wmma_bf16(a[i].v, bfr[j].v, acc[i][j]);
        }
        p ^= 1;
    }
#else
    // Fallback: cooperative vector loads (no TDM builtin available)
    bf16* As = smem;
    bf16* Bs = smem + 2 * ASZ;
    for (int k0 = 0; k0 < K; k0 += 64) {
        __syncthreads();
        for (int e = 0; e < 4; e++) {          // 128x64 bf16 = 1024 uint4 each
            int u = tid + e * 256;
            int r = u >> 3, c = (u & 7) << 3;
            *reinterpret_cast<uint4*>(&As[r * 64 + c]) =
                *reinterpret_cast<const uint4*>(&A[(size_t)(m0 + r) * K + k0 + c]);
            *reinterpret_cast<uint4*>(&Bs[r * 64 + c]) =
                *reinterpret_cast<const uint4*>(&W[(size_t)(n0 + r) * K + k0 + c]);
        }
        __syncthreads();
        for (int kk = 0; kk < 2; kk++) {
            Frag a[4], bfr[2];
            for (int i = 0; i < 4; i++)
                a[i] = ld_afrag(&As[(wm * 64 + i * 16 + l) * 64 + kk * 32], h);
            for (int j = 0; j < 2; j++)
                bfr[j] = ld_bfrag(&Bs[(wn * 32 + j * 16 + l) * 64 + kk * 32], h);
            for (int i = 0; i < 4; i++)
                for (int j = 0; j < 2; j++)
                    acc[i][j] = wmma_bf16(a[i].v, bfr[j].v, acc[i][j]);
        }
    }
#endif

    // Epilogue. C layout: lane-half h, VGPR v -> row v+8h, col = lane&15.
    for (int i = 0; i < 4; i++) {
        for (int j = 0; j < 2; j++) {
            const int col = n0 + wn * 32 + j * 16 + l;
            const float bv = bias[col];
            for (int v = 0; v < 8; v++) {
                const int row = m0 + wm * 64 + i * 16 + v + h * 8;
                const size_t idx = (size_t)row * N + col;
                float xv = acc[i][j][v] + bv;
                if (EPI == 0) {
                    ((bf16*)out)[idx] = (bf16)xv;
                } else if (EPI == 1) {
                    ((float*)out)[idx] = xv + resid[idx];
                } else {
                    float gl = 0.5f * xv * (1.0f + erff(xv * 0.70710678118654752f));
                    ((bf16*)out)[idx] = (bf16)gl;
                }
            }
        }
    }
}

// ---------------------------------------------------------------------------
// Flash attention, causal. qkv bf16 [S,B,3H]; block = 128 query rows of one
// (b, head); 8 waves x 16 rows. K-tile DMA'd by TDM; V staged transposed.
// ---------------------------------------------------------------------------
__global__ __launch_bounds__(256) void attn_kernel(
    const bf16* __restrict__ qkv, bf16* __restrict__ out)
{
    // [kbuf 64x128 | vt 128x64 | pbuf 8x16x64]
    __shared__ bf16 smem[64 * 128 + 128 * 64 + 8 * 16 * 64];
    bf16* kbuf = smem;                  // LDS byte offset 0
    bf16* vt   = smem + 64 * 128;
    bf16* pbuf = smem + 64 * 128 + 128 * 64;

    const int tid = threadIdx.x, lane = tid & 31, wid = tid >> 5;
    const int l = lane & 15, h = lane >> 4;
    const int m0 = blockIdx.x * 128;
    const int nh = blockIdx.y;
    const int b  = blockIdx.z;
    const float scale = 0.08838834764831845f;   // 1/sqrt(128)
    const size_t rstride = (size_t)BATCH * 3 * HDIM;   // elems between seq rows

    Frag q[4];
    {
        const int srow = m0 + wid * 16 + l;
        const bf16* qp = qkv + (size_t)(srow * BATCH + b) * (3 * HDIM) + nh * HEADD;
        for (int c = 0; c < 4; c++) {
            q[c].u[0] = *reinterpret_cast<const uint4*>(qp + c * 32 + h * 8);
            q[c].u[1] = *reinterpret_cast<const uint4*>(qp + c * 32 + h * 8 + 16);
        }
    }

    f32x8 o[8];
    for (int dd = 0; dd < 8; dd++) o[dd] = fzero();
    float rmax[8], rsum[8];
    for (int v = 0; v < 8; v++) { rmax[v] = -1e30f; rsum[v] = 0.f; }

    bf16* pb = &pbuf[wid * 16 * 64];

    for (int t0 = 0; t0 < m0 + 128; t0 += 64) {
        __syncthreads();
        // K tile [t][hd] 64x128
#if HAS_TDM
        if (wid == 0) {
            const bf16* kp = qkv + (size_t)(t0 * BATCH + b) * (3 * HDIM)
                             + HDIM + nh * HEADD;
            tdm_load_2d(0u, kp, 128, 64, rstride);
        }
#else
        for (int e = 0; e < 4; e++) {
            int u = tid + e * 256;
            int r = u >> 4, c = (u & 15) << 3;
            const bf16* src = qkv + (size_t)((t0 + r) * BATCH + b) * (3 * HDIM)
                              + HDIM + nh * HEADD + c;
            *reinterpret_cast<uint4*>(&kbuf[r * 128 + c]) =
                *reinterpret_cast<const uint4*>(src);
        }
#endif
        // V tile transposed into vt[hd][t]
        for (int e = 0; e < 4; e++) {
            int u = tid + e * 256;
            int r = u >> 4, c = (u & 15) << 3;
            const bf16* src = qkv + (size_t)((t0 + r) * BATCH + b) * (3 * HDIM)
                              + 2 * HDIM + nh * HEADD + c;
            uint4 d = *reinterpret_cast<const uint4*>(src);
            const bf16* dv = reinterpret_cast<const bf16*>(&d);
            for (int jj = 0; jj < 8; jj++) vt[(c + jj) * 64 + r] = dv[jj];
        }
#if HAS_TDM
        if (wid == 0) __builtin_amdgcn_s_wait_tensorcnt(0);
#endif
        __syncthreads();

        // Scores: S(16x64) = Q(16x128) * K^T
        f32x8 sf[4];
        for (int j = 0; j < 4; j++) {
            f32x8 accs = fzero();
            for (int c = 0; c < 4; c++) {
                Frag kb = ld_bfrag(&kbuf[(j * 16 + l) * 128 + c * 32], h);
                accs = wmma_bf16(q[c].v, kb.v, accs);
            }
            sf[j] = accs;
        }

        // Scale + causal mask + online softmax stats
        const int rbase = m0 + wid * 16 + h * 8;
        float nmax[8];
        for (int v = 0; v < 8; v++) {
            float tm = -1e30f;
            for (int j = 0; j < 4; j++) {
                float sv = sf[j][v] * scale;
                const int col = t0 + j * 16 + l;
                if (col > rbase + v) sv = -1e30f;
                sf[j][v] = sv;
                tm = fmaxf(tm, sv);
            }
            for (int off = 1; off < 16; off <<= 1)
                tm = fmaxf(tm, __shfl_xor(tm, off, 32));
            nmax[v] = fmaxf(rmax[v], tm);
        }
        for (int v = 0; v < 8; v++) {
            const float alpha = __expf(rmax[v] - nmax[v]);
            rmax[v] = nmax[v];
            float rs = 0.f;
            for (int j = 0; j < 4; j++) {
                float p = __expf(sf[j][v] - nmax[v]);
                rs += p;
                pb[(v + h * 8) * 64 + j * 16 + l] = (bf16)p;   // C-layout -> LDS
            }
            for (int off = 1; off < 16; off <<= 1)
                rs += __shfl_xor(rs, off, 32);
            rsum[v] = rsum[v] * alpha + rs;
            for (int dd = 0; dd < 8; dd++) o[dd][v] *= alpha;
        }

        // O += P(16x64) * V(64x128): P as A-frags, V as B-frags (transposed LDS)
        Frag pa[2];
        for (int kk = 0; kk < 2; kk++)
            pa[kk] = ld_afrag(&pb[l * 64 + kk * 32], h);
        for (int dd = 0; dd < 8; dd++) {
            for (int kk = 0; kk < 2; kk++) {
                Frag vb = ld_bfrag(&vt[(dd * 16 + l) * 64 + kk * 32], h);
                o[dd] = wmma_bf16(pa[kk].v, vb.v, o[dd]);
            }
        }
    }

    // Normalize and write [S,B,H] bf16
    const int rbase = m0 + wid * 16 + h * 8;
    for (int dd = 0; dd < 8; dd++) {
        for (int v = 0; v < 8; v++) {
            const int srow = rbase + v;
            const float val = o[dd][v] / rsum[v];
            out[(size_t)(srow * BATCH + b) * HDIM + nh * HEADD + dd * 16 + l] = (bf16)val;
        }
    }
}

// ---------------------------------------------------------------------------
// Host-side orchestration
// ---------------------------------------------------------------------------
extern "C" void kernel_launch(void* const* d_in, const int* in_sizes, int n_in,
                              void* d_out, int out_size, void* d_ws, size_t ws_size,
                              hipStream_t stream) {
    const float* x      = (const float*)d_in[0];
    const float* ln1_g  = (const float*)d_in[1];
    const float* ln1_b  = (const float*)d_in[2];
    const float* qkv_w  = (const float*)d_in[3];
    const float* qkv_b  = (const float*)d_in[4];
    const float* proj_w = (const float*)d_in[5];
    const float* proj_b = (const float*)d_in[6];
    const float* ln2_g  = (const float*)d_in[7];
    const float* ln2_b  = (const float*)d_in[8];
    const float* fc1_w  = (const float*)d_in[9];
    const float* fc1_b  = (const float*)d_in[10];
    const float* fc2_w  = (const float*)d_in[11];
    const float* fc2_b  = (const float*)d_in[12];
    float* outp = (float*)d_out;

    const size_t MB = 1048576;
    char* w = (char*)d_ws;
    bf16*  ln_buf   = (bf16*) (w + 0);          // 32 MB  bf16 [M,H]
    bf16*  qkv_buf  = (bf16*) (w + 32 * MB);    // 96 MB  bf16 [M,3H]
    bf16*  attn_buf = (bf16*) (w + 128 * MB);   // 32 MB  bf16 [M,H]
    float* x2_buf   = (float*)(w + 160 * MB);   // 64 MB  f32  [M,H]
    bf16*  h3_buf   = (bf16*) (w + 32 * MB);    // 128 MB bf16 [M,4H] (alias qkv+attn)
    bf16*  wqkv     = (bf16*) (w + 224 * MB);   // 24 MB
    bf16*  wproj    = (bf16*) (w + 248 * MB);   //  8 MB
    bf16*  wfc1     = (bf16*) (w + 256 * MB);   // 32 MB
    bf16*  wfc2     = (bf16*) (w + 288 * MB);   // 32 MB   (total 320 MB)

    const dim3 blk(256);

    // 0. one-time weight downconversion f32 -> bf16
    { int n = 3 * HDIM * HDIM; cvt_kernel<<<n / 1024, blk, 0, stream>>>(qkv_w,  wqkv,  n); }
    { int n = HDIM * HDIM;     cvt_kernel<<<n / 1024, blk, 0, stream>>>(proj_w, wproj, n); }
    { int n = 4 * HDIM * HDIM; cvt_kernel<<<n / 1024, blk, 0, stream>>>(fc1_w,  wfc1,  n); }
    { int n = 4 * HDIM * HDIM; cvt_kernel<<<n / 1024, blk, 0, stream>>>(fc2_w,  wfc2,  n); }

    // 1. LN1 -> bf16
    ln_kernel<<<MROWS, blk, 0, stream>>>(x, ln1_g, ln1_b, ln_buf, HDIM);

    // 2. qkv = ln1 @ qkv_w^T + qkv_b  -> bf16 [M, 3H]
    gemm_kernel<0><<<dim3(3 * HDIM / 128, MROWS / 128), blk, 0, stream>>>(
        ln_buf, wqkv, qkv_b, nullptr, qkv_buf, MROWS, 3 * HDIM, HDIM);

    // 3. causal flash attention -> bf16 [M, H]
    attn_kernel<<<dim3(SEQ / 128, NHEADS, BATCH), blk, 0, stream>>>(qkv_buf, attn_buf);

    // 4. x2 = attn @ proj_w^T + proj_b + x   -> f32
    gemm_kernel<1><<<dim3(HDIM / 128, MROWS / 128), blk, 0, stream>>>(
        attn_buf, wproj, proj_b, x, x2_buf, MROWS, HDIM, HDIM);

    // 5. LN2 -> bf16
    ln_kernel<<<MROWS, blk, 0, stream>>>(x2_buf, ln2_g, ln2_b, ln_buf, HDIM);

    // 6. h3 = gelu(ln2 @ fc1_w^T + fc1_b)  -> bf16 [M, 4H]
    gemm_kernel<2><<<dim3(4 * HDIM / 128, MROWS / 128), blk, 0, stream>>>(
        ln_buf, wfc1, fc1_b, nullptr, h3_buf, MROWS, 4 * HDIM, HDIM);

    // 7. out = h3 @ fc2_w^T + fc2_b + x2   -> f32
    gemm_kernel<1><<<dim3(HDIM / 128, MROWS / 128), blk, 0, stream>>>(
        h3_buf, wfc2, fc2_b, x2_buf, outp, MROWS, HDIM, 4 * HDIM);
}
